// graph_directed_sep_63651415327269
// MI455X (gfx1250) — matrix operand order repaired and checked by reading, not compiled, not run on hardware
//
#include <hip/hip_runtime.h>
#include <hip/hip_fp16.h>

// Problem constants (match reference)
#define NM    3
#define NSUB  2000
#define NTOT  (NM * NSUB)          // 6000
#define DIM   64
#define ALPHA 3.0f

#define TILES_PER_DIM      (NSUB / 16)                       // 125
#define TILES_PER_BLOCKMAT (TILES_PER_DIM * TILES_PER_DIM)   // 15625
#define NTILES             (NM * NM * TILES_PER_BLOCKMAT)    // 140625

typedef __attribute__((ext_vector_type(16))) _Float16 v16h;
typedef __attribute__((ext_vector_type(8)))  _Float16 v8h;
typedef __attribute__((ext_vector_type(8)))  float    v8f;

// Hardware tanh (gfx1250 V_TANH_F32) when the builtin exists; libm fallback.
__device__ __forceinline__ float htanh(float x)
{
#if __has_builtin(__builtin_amdgcn_tanhf)
    return __builtin_amdgcn_tanhf(x);
#elif __has_builtin(__builtin_amdgcn_tanh_f32)
    return __builtin_amdgcn_tanh_f32(x);
#else
    return tanhf(x);
#endif
}

// ---------------------------------------------------------------------------
// Phase 0: nv[kb,n,e] = tanh(ALPHA * (emb[kb,n,:] . W[kb,e,:] + b[kb,e]))
// stored as f16 (WMMA operand precision). 9*2000*64 outputs, 64 MACs each.
// ---------------------------------------------------------------------------
__global__ __launch_bounds__(256)
void nv_kernel(const float* __restrict__ emb, const float* __restrict__ W,
               const float* __restrict__ b, _Float16* __restrict__ nv)
{
    int t = blockIdx.x * blockDim.x + threadIdx.x;
    if (t >= NM * NM * NSUB * DIM) return;
    int e  = t & (DIM - 1);
    int n  = (t / DIM) % NSUB;
    int kb = t / (DIM * NSUB);

    const float* er = emb + (size_t)(kb * NSUB + n) * DIM;
    const float* wr = W   + (size_t)(kb * DIM  + e) * DIM;
    float acc = b[kb * DIM + e];
#pragma unroll
    for (int d = 0; d < DIM; ++d) acc = fmaf(er[d], wr[d], acc);
    nv[t] = (_Float16)htanh(ALPHA * acc);
}

// ---------------------------------------------------------------------------
// Phase 1: adjacency via WMMA. One wave32 per 16x16 output tile, K=64 done as
// two v_wmma_f32_16x16x32_f16. Epilogue relu(tanh(ALPHA*x)) in f32.
//
// A operand (16x32 f16, contiguous k order chosen here, same as B; not the ISA table order): lane L -> row M=L%16, half h=L/16;
//   a[0..15] = K (k0 + 16h + 0..15)  == contiguous 32B of nv1 row M.
// B operand (32x16 f16) with B = nv2^T: lane L -> col N=L%16, half h=L/16;
//   b[0..15] = K (k0 + 16h + 0..15)  == contiguous 32B of nv2 row N.
// D (16x16 f32): acc[r] -> row M = r + 8h, col N = L%16.
// ---------------------------------------------------------------------------
__global__ __launch_bounds__(256)
void adj_kernel(const _Float16* __restrict__ nv1,
                const _Float16* __restrict__ nv2,
                float* __restrict__ out)
{
    int wave = blockIdx.x * (blockDim.x >> 5) + (threadIdx.x >> 5);
    if (wave >= NTILES) return;              // wave-uniform: EXEC all-1s inside
    int lane = threadIdx.x & 31;
    int half = lane >> 4;
    int lmod = lane & 15;

    int kb = wave / TILES_PER_BLOCKMAT;
    int r  = wave % TILES_PER_BLOCKMAT;
    int tm = r / TILES_PER_DIM;
    int tn = r % TILES_PER_DIM;
    int bi = kb / NM;                        // row block of the 3x3 layout
    int bj = kb % NM;                        // col block

    const _Float16* Abase = nv1 + (size_t)(kb * NSUB + tm * 16 + lmod) * DIM;
    const _Float16* Bbase = nv2 + (size_t)(kb * NSUB + tn * 16 + lmod) * DIM;

    v8f acc = {};
#pragma unroll
    for (int k0 = 0; k0 < DIM; k0 += 32) {
        v16h a = *(const v16h*)(Abase + k0 + 16 * half);
        v16h bm = *(const v16h*)(Bbase + k0 + 16 * half);
        acc = __builtin_amdgcn_wmma_f32_16x16x32_f16(
                  /*neg_a=*/false, a, /*neg_b=*/false, bm,
                  /*c_mod=*/(short)0, acc, /*reuse_a=*/false, /*reuse_b=*/false);
    }

    size_t rowbase = (size_t)(bi * NSUB + tm * 16);
    size_t col     = (size_t)(bj * NSUB + tn * 16 + lmod);
#pragma unroll
    for (int rr = 0; rr < 8; ++rr) {
        float v = htanh(ALPHA * acc[rr]);
        v = v > 0.0f ? v : 0.0f;             // relu
        size_t row = rowbase + rr + 8 * half;
        out[row * (size_t)NTOT + col] = v;   // 16-lane contiguous 64B segments
    }
}

// ---------------------------------------------------------------------------
// Phase 2: in-place row-wise top-k. One block per row; row cached in LDS.
// Exact k-th largest via 31-step bitwise binary search on float bit patterns
// (all values >= 0 after relu, so uint order == float order).
// ---------------------------------------------------------------------------
__global__ __launch_bounds__(256)
void topk_kernel(float* __restrict__ out, const int* __restrict__ kptr)
{
    __shared__ float srow[NTOT];
    __shared__ int   scount;
    const int row = blockIdx.x;
    const int tid = threadIdx.x;
    const int K   = *kptr;

    float* rp = out + (size_t)row * NTOT;
    for (int i = tid; i < NTOT; i += 256) srow[i] = rp[i];
    __syncthreads();

    unsigned p = 0u;
    for (int bit = 30; bit >= 0; --bit) {
        unsigned cand = p | (1u << bit);
        if (tid == 0) scount = 0;
        __syncthreads();
        int c = 0;
        for (int i = tid; i < NTOT; i += 256)
            c += (__float_as_uint(srow[i]) >= cand) ? 1 : 0;
        atomicAdd(&scount, c);
        __syncthreads();
        if (scount >= K) p = cand;           // largest p with count(x>=p) >= K
        __syncthreads();
    }
    const float thr = __uint_as_float(p);    // == K-th largest value

    for (int i = tid; i < NTOT; i += 256) {
        float v = srow[i];
        rp[i] = (v >= thr) ? v : 0.0f;
    }
}

// ---------------------------------------------------------------------------
extern "C" void kernel_launch(void* const* d_in, const int* in_sizes, int n_in,
                              void* d_out, int out_size, void* d_ws, size_t ws_size,
                              hipStream_t stream)
{
    // inputs: 0=idx (unused), 1=emb1, 2=emb2, 3=W1, 4=b1, 5=W2, 6=b2, 7=k
    const float* emb1 = (const float*)d_in[1];
    const float* emb2 = (const float*)d_in[2];
    const float* W1   = (const float*)d_in[3];
    const float* b1   = (const float*)d_in[4];
    const float* W2   = (const float*)d_in[5];
    const float* b2   = (const float*)d_in[6];
    const int*   kptr = (const int*)d_in[7];
    float* out = (float*)d_out;

    _Float16* nv1 = (_Float16*)d_ws;                          // 9*2000*64 f16
    _Float16* nv2 = nv1 + (size_t)NM * NM * NSUB * DIM;       // + same

    const int nvElems  = NM * NM * NSUB * DIM;                // 1,152,000
    const int nvBlocks = (nvElems + 255) / 256;
    nv_kernel<<<nvBlocks, 256, 0, stream>>>(emb1, W1, b1, nv1);
    nv_kernel<<<nvBlocks, 256, 0, stream>>>(emb2, W2, b2, nv2);

    const int adjBlocks = (NTILES + 7) / 8;                   // 8 waves / block
    adj_kernel<<<adjBlocks, 256, 0, stream>>>(nv1, nv2, out);

    topk_kernel<<<NTOT, 256, 0, stream>>>(out, kptr);
}
